// PositionAttention_21792664060049
// MI455X (gfx1250) — matrix operand order repaired and checked by reading, not compiled, not run on hardware
//
#include <hip/hip_runtime.h>
#include <hip/hip_bf16.h>
#include <math.h>

// ---------------------------------------------------------------------------
// PositionAttention:  Bp = WB @ x ;  E = attention(Q=K=V=Bp^T) ;
// out = gamma*E + x.   B=4, C=64, H=W=64 -> N=4096, head dim d=64.
// Flash-attention with v_wmma_f32_16x16x32_bf16, online softmax, S never
// materialized.  Computes S^T = K Q^T so softmax stats reduce within lanes
// (+ one half-wave shfl), and O^T = V^T S^T so output lands in [b][c][n].
// ---------------------------------------------------------------------------

typedef __bf16 bf16_t;
typedef __attribute__((ext_vector_type(16))) __bf16 v16bf;
typedef __attribute__((ext_vector_type(8)))  float  v8f;
typedef __attribute__((ext_vector_type(4)))  unsigned int u32x4;

#define B_   4
#define C_   64        // channels == head dim
#define NPOS 4096      // H*W

union FragU { u32x4 u[2]; v16bf v; };

// 16 contiguous bf16 (one A-operand lane slice: a V^T row slice)
static __device__ __forceinline__ v16bf load_frag_c(const bf16_t* p) {
  FragU f;
  f.u[0] = *(const u32x4*)(p);
  f.u[1] = *(const u32x4*)(p + 8);
  return f.v;
}
// two separate 8-bf16 chunks (A-operand lane slice)
static __device__ __forceinline__ v16bf load_frag_2(const bf16_t* pa, const bf16_t* pb) {
  FragU f;
  f.u[0] = *(const u32x4*)(pa);
  f.u[1] = *(const u32x4*)(pb);
  return f.v;
}

static __device__ __forceinline__ v8f wmma_bf16(v16bf a, v16bf b, v8f c) {
  // D(f32 16x16) = A(bf16 16x32) * B(bf16 32x16) + C
  return __builtin_amdgcn_wmma_f32_16x16x32_bf16(false, a, false, b, (short)0, c,
                                                 false, false);
}

// ---------------------------------------------------------------------------
// Kernel 1: Bp = WB @ x, emitted twice in bf16:
//   Pq[b][n][c]  (row-major positions -> Q/K operand fragments)
//   Pt[b][c][n]  (natural Bp layout   -> V^T operand fragments)
// ---------------------------------------------------------------------------
__global__ __launch_bounds__(256) void pa_conv1x1(const float* __restrict__ x,
                                                  const float* __restrict__ WB,
                                                  bf16_t* __restrict__ Pq,
                                                  bf16_t* __restrict__ Pt) {
  __shared__ float wb[C_ * C_];
  const int tid = threadIdx.x;
  for (int i = tid; i < C_ * C_; i += 256) wb[i] = WB[i];
  __syncthreads();

  const int blk = blockIdx.x;              // B_ * (NPOS/64) blocks
  const int b   = blk / (NPOS / 64);
  const int n0  = (blk % (NPOS / 64)) * 64;
  const int p   = n0 + (tid & 63);         // position
  const int cg  = tid >> 6;                // channel group 0..3 (16 ch each)

  const float* xb = x + (size_t)b * C_ * NPOS;

  float acc[16];
#pragma unroll
  for (int i = 0; i < 16; ++i) acc[i] = 0.f;

  for (int k = 0; k < C_; ++k) {
    const float xv = xb[(size_t)k * NPOS + p];
#pragma unroll
    for (int i = 0; i < 16; ++i)
      acc[i] += wb[(cg * 16 + i) * C_ + k] * xv;
  }

  bf16_t* pq = Pq + ((size_t)b * NPOS + p) * C_ + cg * 16;
#pragma unroll
  for (int i = 0; i < 16; ++i) pq[i] = (bf16_t)acc[i];

  bf16_t* pt = Pt + ((size_t)b * C_ + cg * 16) * NPOS + p;
#pragma unroll
  for (int i = 0; i < 16; ++i) pt[(size_t)i * NPOS] = (bf16_t)acc[i];
}

// ---------------------------------------------------------------------------
// Kernel 2: flash attention.  One wave owns 16 queries; 8 waves / block.
// grid = B_*NPOS/(16*8) = 128 blocks.
// ---------------------------------------------------------------------------
__global__ __launch_bounds__(256) void pa_flash_attn(const bf16_t* __restrict__ Pq,
                                                     const bf16_t* __restrict__ Pt,
                                                     const float* __restrict__ x,
                                                     const float* __restrict__ gamma,
                                                     float* __restrict__ out) {
  const int lane  = threadIdx.x & 31;
  const int wave  = threadIdx.x >> 5;
  const int qtile = blockIdx.x * 8 + wave;      // 0 .. B_*NPOS/16 - 1
  const int b     = qtile / (NPOS / 16);
  const int q0    = (qtile % (NPOS / 16)) * 16;
  const int lmod  = lane & 15;
  const int lhalf = lane >> 4;                  // 0: lanes 0-15, 1: lanes 16-31

  const bf16_t* PqB = Pq + (size_t)b * NPOS * C_;
  const bf16_t* PtB = Pt + (size_t)b * C_ * NPOS;

  // Resident Q^T B-operand fragments: B[k][n] = Q[q0+n][feat-chunk + k].
  // B layout: lane holds column n = lane%16, two 8-wide K (feature) chunks
  // (k 8h..8h+7 and 16+8h..16+8h+7) selected by lane half h.
  const bf16_t* qrow = PqB + (size_t)(q0 + lmod) * C_;
  const v16bf qb0 = load_frag_2(qrow + 0 * 32 + lhalf * 8, qrow + 0 * 32 + 16 + lhalf * 8);   // features 0..31
  const v16bf qb1 = load_frag_2(qrow + 1 * 32 + lhalf * 8, qrow + 1 * 32 + 16 + lhalf * 8);   // features 32..63

  v8f acc[4];                                   // O^T: 4 tiles of (16 feat x 16 q)
#pragma unroll
  for (int f = 0; f < 4; ++f) acc[f] = (v8f){};
  float m = -INFINITY;
  float l = 0.f;

  for (int k0 = 0; k0 < NPOS; k0 += 32) {
    // --- GEMM1: S^T tiles  (A = key rows from Pq, K-dim = features) -------
    const bf16_t* krow0 = PqB + (size_t)(k0 + lmod) * C_;
    const bf16_t* krow1 = PqB + (size_t)(k0 + 16 + lmod) * C_;
    __builtin_prefetch(krow0 + 32 * C_, 0, 1);

    const v16bf ka00 = load_frag_2(krow0 +      lhalf * 8, krow0 + 16 + lhalf * 8);
    const v16bf ka01 = load_frag_2(krow0 + 32 + lhalf * 8, krow0 + 48 + lhalf * 8);
    const v16bf ka10 = load_frag_2(krow1 +      lhalf * 8, krow1 + 16 + lhalf * 8);
    const v16bf ka11 = load_frag_2(krow1 + 32 + lhalf * 8, krow1 + 48 + lhalf * 8);

    v8f s0 = (v8f){};                    // keys k0..k0+15   x queries
    s0 = wmma_bf16(ka00, qb0, s0);
    s0 = wmma_bf16(ka01, qb1, s0);
    v8f s1 = (v8f){};                    // keys k0+16..k0+31 x queries
    s1 = wmma_bf16(ka10, qb0, s1);
    s1 = wmma_bf16(ka11, qb1, s1);

    // --- online softmax over the 32 keys of this chunk --------------------
    // lane holds C/D rows r + lhalf*8 of each tile, column q = lane%16.
    float lmax = -INFINITY;
#pragma unroll
    for (int r = 0; r < 8; ++r) {
      lmax = fmaxf(lmax, s0[r]);
      lmax = fmaxf(lmax, s1[r]);
    }
    lmax = fmaxf(lmax, __shfl_xor(lmax, 16, 32));   // other half-wave: other rows
    const float mnew  = fmaxf(m, lmax);
    const float alpha = __expf(m - mnew);           // 0 on first iteration
    m = mnew;

    float p0[8], p1[8];
    float rs = 0.f;
#pragma unroll
    for (int r = 0; r < 8; ++r) {
      p0[r] = __expf(s0[r] - mnew);
      p1[r] = __expf(s1[r] - mnew);
      rs += p0[r] + p1[r];
    }
    rs += __shfl_xor(rs, 16, 32);
    l = l * alpha + rs;

    // --- reshape exp(S^T) into B operand (32 keys x 16 queries, bf16) -----
    // K elems 0..15 (lanes<16): tile0 rows 0..7 own, rows 8..15 from partner.
    // K elems 16..31 (lanes>=16): tile1 rows 0..7 from partner, 8..15 own.
    v16bf pb;
#pragma unroll
    for (int r = 0; r < 8; ++r) {
      const float x0 = __shfl_xor(p0[r], 16, 32);
      const float x1 = __shfl_xor(p1[r], 16, 32);
      const float e_lo = lhalf ? x1    : p0[r];
      const float e_hi = lhalf ? p1[r] : x0;
      pb[r]     = (bf16_t)e_lo;
      pb[r + 8] = (bf16_t)e_hi;
    }

    // --- rescale accumulators, then O^T += V^T * exp(S^T) -----------------
#pragma unroll
    for (int f = 0; f < 4; ++f) {
#pragma unroll
      for (int r = 0; r < 8; ++r) acc[f][r] *= alpha;
    }
#pragma unroll
    for (int f = 0; f < 4; ++f) {
      // A = V^T: 16 features x 32 keys, rows from Pt (feature-major)
      const bf16_t* vrow = PtB + (size_t)(f * 16 + lmod) * NPOS + k0;
      const v16bf va = load_frag_c(vrow + lhalf * 16);
      acc[f] = wmma_bf16(va, pb, acc[f]);
    }
  }

  // --- epilogue: out[b][c][n] = gamma * O^T[c][q]/l + x[b][c][n] ----------
  const float g    = gamma[0];
  const float ginv = g / l;
  const float* xb = x   + (size_t)b * C_ * NPOS;
  float*       ob = out + (size_t)b * C_ * NPOS;
  const int q = q0 + lmod;
#pragma unroll
  for (int f = 0; f < 4; ++f) {
#pragma unroll
    for (int r = 0; r < 8; ++r) {
      const int c = f * 16 + lhalf * 8 + r;
      const size_t idx = (size_t)c * NPOS + q;
      ob[idx] = acc[f][r] * ginv + xb[idx];
    }
  }
}

// ---------------------------------------------------------------------------
extern "C" void kernel_launch(void* const* d_in, const int* in_sizes, int n_in,
                              void* d_out, int out_size, void* d_ws, size_t ws_size,
                              hipStream_t stream) {
  const float* x     = (const float*)d_in[0];   // [4,64,64,64]
  const float* WB    = (const float*)d_in[1];   // [64,64]
  const float* gamma = (const float*)d_in[2];   // [1]
  float*       out   = (float*)d_out;

  bf16_t* Pq = (bf16_t*)d_ws;                               // [B][N][C] bf16, 2 MB
  bf16_t* Pt = Pq + (size_t)B_ * NPOS * C_;                 // [B][C][N] bf16, 2 MB

  pa_conv1x1<<<B_ * (NPOS / 64), 256, 0, stream>>>(x, WB, Pq, Pt);
  pa_flash_attn<<<(B_ * NPOS / 16) / 8, 256, 0, stream>>>(Pq, Pt, x, gamma, out);
}
